// InteractionModule_7645041786958
// MI455X (gfx1250) — matrix operand order repaired and checked by reading, not compiled, not run on hardware
//
#include <hip/hip_runtime.h>

// ---------------------------------------------------------------------------
// E(3)-equivariant GNN forward for MI455X (gfx1250, wave32, WMMA).
// Dominant cost: per-edge FC GEMMs (~59 GFLOP) -> fused f16 WMMA
// (v_wmma_f32_16x16x32_f16) on 32-edge wave tiles (two 16x16 M-tiles per
// B-fragment load to halve L2 weight traffic); GEMM2 A-fragments hoisted to
// registers so the N-tile hot loop is pure global_load_b128 + v_wmma.
// Tensor-product (CG) consumption fused in LDS so edge weights never hit HBM.
// Workspace use (~21 MB): sh + e_emb + h + sum + cnt + packed f16 weights.
// ---------------------------------------------------------------------------

typedef _Float16 v8h  __attribute__((ext_vector_type(8)));
typedef _Float16 v16h __attribute__((ext_vector_type(16)));
typedef float    v8f  __attribute__((ext_vector_type(8)));

constexpr int NN = 10000;     // nodes
constexpr int NEG = 80000;    // edges
constexpr int HD = 80;        // padded node feature width (final out_dim)
constexpr int EF = 96;        // EDGE_FEAT
constexpr float EPSV = 1e-5f;

// ---- per-layer tensor-product path metadata (hand-derived from build_paths)
struct Path { int ws, wn, in_s, out_s, m3, ty; float alpha; };
// ty: 0=(l1,l2,l3)=(0,0,0) 1=(0,1,1) 2=(1,0,1) 3=(1,1,0) 4=(1,1,1) 5=(1,2,1)

template<int L> struct Cfg;
template<> struct Cfg<0> {
  static constexpr int WN=384, NT2=24, OD=40, NP=2, NIR=2;
  static constexpr Path P[NP] = {
    {   0,256, 0, 0,16,0, 0.25000000f},
    { 256,128, 0,16, 8,1, 0.43301270f},
  };
  static constexpr int IRR[NIR][3] = {{16,0,1},{8,1,-1}};
};
template<> struct Cfg<1> {
  static constexpr int WN=704, NT2=44, OD=64, NP=6, NIR=3;
  static constexpr Path P[NP] = {
    {   0,256, 0, 0,16,0, 0.20412415f},
    { 256,128, 0,16, 8,1, 0.30618622f},
    { 384, 64,16,16, 8,2, 0.30618622f},
    { 448,128,16, 0,16,3, 0.20412415f},
    { 576, 64,16,40, 8,4, 0.61237244f},
    { 640, 64,16,16, 8,5, 0.30618622f},
  };
  static constexpr int IRR[NIR][3] = {{16,0,1},{8,1,-1},{8,1,1}};
};
template<> struct Cfg<2> {
  static constexpr int WN=1024, NT2=64, OD=80, NP=10, NIR=4;
  static constexpr Path P[NP] = {
    {   0,256, 0, 0,16,0, 0.20412415f},
    { 256,128, 0,16, 8,1, 0.27386128f},
    { 384, 64,16,16, 8,2, 0.27386128f},
    { 448,128,16, 0,16,3, 0.20412415f},
    { 576, 64,16,40, 8,4, 0.35355339f},
    { 640, 64,16,16, 8,5, 0.27386128f},
    { 704, 64,40,40, 8,2, 0.35355339f},
    { 768, 64,40,16, 8,4, 0.27386128f},
    { 832,128,40,64,16,3, 0.35355339f},
    { 960, 64,40,40, 8,5, 0.35355339f},
  };
  static constexpr int IRR[NIR][3] = {{16,0,1},{8,1,-1},{8,1,1},{16,0,-1}};
};
template<> struct Cfg<3> {
  static constexpr int WN=1368, NT2=86, OD=80, NP=12, NIR=4;
  static constexpr Path P[NP] = {
    {   0,256, 0, 0,16,0, 0.20412415f},
    { 256,128, 0,16, 8,1, 0.27386128f},
    { 384, 64,16,16, 8,2, 0.27386128f},
    { 448,128,16, 0,16,3, 0.20412415f},
    { 576, 64,16,40, 8,4, 0.27386128f},
    { 640, 64,16,16, 8,5, 0.27386128f},
    { 704, 64,40,40, 8,2, 0.27386128f},
    { 768, 64,40,16, 8,4, 0.27386128f},
    { 832,128,40,64,16,3, 0.20412415f},
    { 960, 64,40,40, 8,5, 0.27386128f},
    {1024,256,64,64,16,0, 0.20412415f},
    {1280,128,64,40, 8,1, 0.27386128f},
  };
  static constexpr int IRR[NIR][3] = {{16,0,1},{8,1,-1},{8,1,1},{16,0,-1}};
};

// ---------------------------------------------------------------------------
__device__ inline void wave_lds_sync() {
#if defined(__AMDGCN__)
  __builtin_amdgcn_wave_barrier();
  asm volatile("s_wait_dscnt 0" ::: "memory");
  __builtin_amdgcn_wave_barrier();
#endif
}

// A-fragment (16x32 f16, M rows) from LDS row-major [rows x EF] tile.
// ISA layout: lanes 0-15 row m, K {k0..k0+7, k0+16..k0+23}; lanes 16-31 +8.
__device__ inline v16h load_a_frag(const _Float16* base, int k0, int lane) {
  const int m = lane & 15, g = lane >> 4;
  v8h lo = *(const v8h*)(base + m*EF + k0 + g*8);
  v8h hi = *(const v8h*)(base + m*EF + k0 + 16 + g*8);
  v16h a;
#pragma unroll
  for (int t = 0; t < 8; ++t) { a[t] = lo[t]; a[8+t] = hi[t]; }
  return a;
}

__device__ inline v8f zero_v8f() { v8f z = {0.f,0.f,0.f,0.f,0.f,0.f,0.f,0.f}; return z; }

// ---------------------------------------------------------------------------
__global__ void init_kernel(const float* __restrict__ na, float* __restrict__ h,
                            float* __restrict__ sum, float* __restrict__ cnt) {
  int i = blockIdx.x * blockDim.x + threadIdx.x;
  if (i < NN * HD) {
    int n = i / HD, d = i - n * HD;
    h[i]   = (d < 16) ? na[n*16 + d] : 0.f;
    sum[i] = 0.f;
  }
  if (i < NN) cnt[i] = 0.f;
}

// edge geometry + spherical harmonics + gaussian-embedding MLP (scalar VALU)
__global__ void geom_kernel(const float* __restrict__ coords, const int* __restrict__ edges,
                            const float* __restrict__ w1, const float* __restrict__ b1,
                            const float* __restrict__ w2, const float* __restrict__ b2,
                            float* __restrict__ sh9, float* __restrict__ eemb,
                            float* __restrict__ cnt) {
  int e = blockIdx.x * blockDim.x + threadIdx.x;
  if (e >= NEG) return;
  int s = edges[e], d = edges[NEG + e];
  float ex = coords[d*3+0]-coords[s*3+0];
  float ey = coords[d*3+1]-coords[s*3+1];
  float ez = coords[d*3+2]-coords[s*3+2];
  float r  = sqrtf(ex*ex + ey*ey + ez*ez + 1e-12f);
  float ux = ex/r, uy = ey/r, uz = ez/r;
  const float S3 = 1.7320508f, S15 = 3.8729833f, S5 = 2.2360680f;
  float sh[9] = {1.f, S3*uy, S3*uz, S3*ux, S15*ux*uy, S15*uy*uz,
                 0.5f*S5*(3.f*uz*uz - 1.f), S15*ux*uz, 0.5f*S15*(ux*ux - uy*uy)};
#pragma unroll
  for (int j = 0; j < 9; ++j) sh9[e*9 + j] = sh[j];
  const float GSTEP = 5.0f / 31.0f;
  const float coeff = -0.5f / (GSTEP * GSTEP);
  float g[32], hid[32];
#pragma unroll
  for (int j = 0; j < 32; ++j) { float dr = r - GSTEP*j; g[j] = __expf(coeff*dr*dr); }
  for (int o = 0; o < 32; ++o) {
    float a = b1[o];
#pragma unroll
    for (int j = 0; j < 32; ++j) a += g[j] * w1[j*32 + o];
    hid[o] = a > 0.f ? a : 0.f;
  }
  for (int o = 0; o < 32; ++o) {
    float a = b2[o];
#pragma unroll
    for (int j = 0; j < 32; ++j) a += hid[j] * w2[j*32 + o];
    eemb[e*32 + o] = a;
  }
  atomicAdd(&cnt[s], 1.0f);
}

// pack f32 row-major W[K x Nvalid] -> f16 WMMA B-fragment tiles
// tile(kt,nt): 32 lanes x 16 halfs; lane n-col = nt*16+(lane&15),
// K = kt*32 + (lane>>4)*8 + t (t<8), kt*32 + 16 + (lane>>4)*8 + (t-8) (t>=8)  -- ISA 16-bit B layout.
__global__ void pack_kernel(const float* __restrict__ W, unsigned short* __restrict__ outp,
                            int K, int Nvalid, int NT) {
  int idx = blockIdx.x * blockDim.x + threadIdx.x;
  int total = (K/32) * NT * 512;
  if (idx >= total) return;
  int tile = idx >> 9, within = idx & 511;
  int lane = within >> 4, t = within & 15;
  int kt = tile / NT, nt = tile - kt*NT;
  int n = nt*16 + (lane & 15);
  int k = kt*32 + (t < 8 ? (lane >> 4)*8 + t : 16 + (lane >> 4)*8 + (t - 8));
  float v = (n < Nvalid) ? W[k*Nvalid + n] : 0.f;
  _Float16 hv = (_Float16)v;
  outp[idx] = *(unsigned short*)&hv;
}

// ---------------------------------------------------------------------------
// Fused per-layer edge kernel: ONE wave per 32-edge tile (single-wave
// workgroup; barrier-free, wave-local LDS sync only).
// ea(32x96) -> WMMA GEMM1 (96x96) -> relu -> WMMA GEMM2 (96xWN, N tiled by 16,
// A-fragments held in registers) -> CG tensor-product consume -> atomic
// scatter into node sums (L2-resident).
template<int L>
__global__ __launch_bounds__(32)
void edge_kernel(const int* __restrict__ edges,
                 const float* __restrict__ eemb, const float* __restrict__ eattr,
                 const float* __restrict__ sh9,  const float* __restrict__ h,
                 const unsigned short* __restrict__ w1p, const unsigned short* __restrict__ w2p,
                 const float* __restrict__ b1,   const float* __restrict__ b2,
                 float* __restrict__ nsum) {
  constexpr int NT2 = Cfg<L>::NT2;
  constexpr int WN  = Cfg<L>::WN;
  constexpr int OD  = Cfg<L>::OD;
  constexpr int ME  = 32;        // edges per wave tile (2 WMMA M-tiles)

  __shared__ alignas(32) _Float16 s_ea [ME*EF];
  __shared__ alignas(32) _Float16 s_hid[ME*EF];
  __shared__ float s_hd [ME*HD];
  __shared__ float s_sh [ME*9];
  __shared__ float s_wt [ME*16];
  __shared__ float s_out[ME*HD];
  __shared__ int   s_src[ME];
  __shared__ int   s_dst[ME];

  const int lane = threadIdx.x & 31;
  const int e0   = blockIdx.x * ME;
  const int g    = lane >> 4;
  const int lm   = lane & 15;

  s_src[lane] = edges[e0 + lane];
  s_dst[lane] = edges[NEG + e0 + lane];
  wave_lds_sync();

  // stage ea = [e_emb | edge_type_attr | h[src,:16] | h[dst,:16]] as f16
  for (int idx = lane; idx < ME*EF; idx += 32) {
    int r = idx / EF, c = idx - r*EF;
    float v;
    if (c < 32)       v = eemb [(e0+r)*32 + c];
    else if (c < 64)  v = eattr[(e0+r)*32 + (c-32)];
    else if (c < 80)  v = h[s_src[r]*HD + (c-64)];
    else              v = h[s_dst[r]*HD + (c-80)];
    s_ea[r*EF + c] = (_Float16)v;
  }
  for (int idx = lane; idx < ME*HD; idx += 32)
    s_hd[idx] = h[s_dst[idx/HD]*HD + (idx % HD)];
  for (int idx = lane; idx < ME*9; idx += 32)
    s_sh[idx] = sh9[(e0 + idx/9)*9 + (idx % 9)];
  for (int idx = lane; idx < ME*HD; idx += 32)
    s_out[idx] = 0.f;
  wave_lds_sync();

  // GEMM1: hidden = relu(ea @ W1 + b1), M=32 N=96 K=96 (2 M-tiles)
  v8f acc[2][6];
#pragma unroll
  for (int m = 0; m < 2; ++m)
#pragma unroll
    for (int nt = 0; nt < 6; ++nt) acc[m][nt] = zero_v8f();
#pragma unroll
  for (int kt = 0; kt < 3; ++kt) {
    v16h a0 = load_a_frag(&s_ea[0],      kt*32, lane);
    v16h a1 = load_a_frag(&s_ea[16*EF],  kt*32, lane);
#pragma unroll
    for (int nt = 0; nt < 6; ++nt) {
      v16h b = *(const v16h*)((const _Float16*)w1p + ((kt*6 + nt)*32 + lane)*16);
      acc[0][nt] = __builtin_amdgcn_wmma_f32_16x16x32_f16(false, a0, false, b, (short)0,
                                                          acc[0][nt], false, false);
      acc[1][nt] = __builtin_amdgcn_wmma_f32_16x16x32_f16(false, a1, false, b, (short)0,
                                                          acc[1][nt], false, false);
    }
  }
#pragma unroll
  for (int nt = 0; nt < 6; ++nt) {
    int n = nt*16 + lm;
    float bias = b1[n];
#pragma unroll
    for (int m = 0; m < 2; ++m)
#pragma unroll
      for (int j = 0; j < 8; ++j) {    // D layout: lane holds col n, rows j+8*g
        float v = acc[m][nt][j] + bias;
        s_hid[(m*16 + j + 8*g)*EF + n] = (_Float16)(v > 0.f ? v : 0.f);
      }
  }
  wave_lds_sync();

  // Hoist GEMM2 A-fragments (depend only on kt) into registers once.
  v16h afr[2][3];
#pragma unroll
  for (int kt = 0; kt < 3; ++kt) {
    afr[0][kt] = load_a_frag(&s_hid[0],     kt*32, lane);
    afr[1][kt] = load_a_frag(&s_hid[16*EF], kt*32, lane);
  }

  // GEMM2 tiled by 16 output cols, consume each w-tile immediately.
  const float* x    = &s_hd [lane*HD];
  const float* sv   = &s_sh [lane*9];
  float*       orow = &s_out[lane*HD];
  for (int nt = 0; nt < NT2; ++nt) {
    v8f a2[2] = {zero_v8f(), zero_v8f()};
#pragma unroll
    for (int kt = 0; kt < 3; ++kt) {
      v16h b = *(const v16h*)((const _Float16*)w2p + ((kt*NT2 + nt)*32 + lane)*16);
      a2[0] = __builtin_amdgcn_wmma_f32_16x16x32_f16(false, afr[0][kt], false, b, (short)0,
                                                     a2[0], false, false);
      a2[1] = __builtin_amdgcn_wmma_f32_16x16x32_f16(false, afr[1][kt], false, b, (short)0,
                                                     a2[1], false, false);
    }
    int n = nt*16 + lm;
    float bias2 = (n < WN) ? b2[n] : 0.f;
#pragma unroll
    for (int m = 0; m < 2; ++m)
#pragma unroll
      for (int j = 0; j < 8; ++j)
        s_wt[(m*16 + j + 8*g)*16 + lm] = a2[m][j] + bias2;
    wave_lds_sync();

    // lane == edge; consume the 16 cols of this tile for its edge
    for (int t = 0; t < 16; ++t) {
      int c = nt*16 + t;
      if (c >= WN) break;
      float wv = s_wt[lane*16 + t];
#pragma unroll
      for (int pi = 0; pi < Cfg<L>::NP; ++pi) {
        const Path P = Cfg<L>::P[pi];
        if (c >= P.ws && c < P.ws + P.wn) {
          int off = c - P.ws;
          int u = off / P.m3, v = off - u*P.m3;
          float aw = P.alpha * wv;
          if (P.ty == 0) {                                // (0,0,0)
            orow[P.out_s + v] += aw * x[P.in_s + u] * sv[0];
          } else if (P.ty == 1) {                         // (0,1,1): delta/sqrt3
            float x0 = x[P.in_s + u] * 0.57735027f;
            orow[P.out_s + v*3 + 0] += aw * x0 * sv[1];
            orow[P.out_s + v*3 + 1] += aw * x0 * sv[2];
            orow[P.out_s + v*3 + 2] += aw * x0 * sv[3];
          } else if (P.ty == 2) {                         // (1,0,1): delta/sqrt3
            const float* xv = &x[P.in_s + u*3];
            float s0 = sv[0] * 0.57735027f;
            orow[P.out_s + v*3 + 0] += aw * xv[0] * s0;
            orow[P.out_s + v*3 + 1] += aw * xv[1] * s0;
            orow[P.out_s + v*3 + 2] += aw * xv[2] * s0;
          } else if (P.ty == 3) {                         // (1,1,0): dot/sqrt3
            const float* xv = &x[P.in_s + u*3];
            float y = (xv[0]*sv[1] + xv[1]*sv[2] + xv[2]*sv[3]) * 0.57735027f;
            orow[P.out_s + v] += aw * y;
          } else if (P.ty == 4) {                         // (1,1,1): cross/sqrt6
            const float* xv = &x[P.in_s + u*3];
            const float C = 0.40824829f;
            orow[P.out_s + v*3 + 0] += aw * C * (xv[1]*sv[3] - xv[2]*sv[2]);
            orow[P.out_s + v*3 + 1] += aw * C * (xv[2]*sv[1] - xv[0]*sv[3]);
            orow[P.out_s + v*3 + 2] += aw * C * (xv[0]*sv[2] - xv[1]*sv[1]);
          } else {                                        // (1,2,1): sym-traceless
            const float* xv = &x[P.in_s + u*3];
            const float* q  = &sv[4];   // (xy, yz, 3z^2-1, xz, x^2-y^2)
            const float A = 0.15811388f, B = 0.12909944f; // ~ wigner_3j(1,2,1)
            float y0 = (-A*q[4] - B*q[2])*xv[0] + A*q[1]*xv[1] + A*q[0]*xv[2];
            float y1 =   A*q[1]*xv[0] + 2.f*B*q[2]*xv[1] + A*q[3]*xv[2];
            float y2 =   A*q[0]*xv[0] + A*q[3]*xv[1] + (A*q[4] - B*q[2])*xv[2];
            orow[P.out_s + v*3 + 0] += aw * y0;
            orow[P.out_s + v*3 + 1] += aw * y1;
            orow[P.out_s + v*3 + 2] += aw * y2;
          }
        }
      }
    }
    wave_lds_sync();
  }

  // scatter messages into node sums (10000x80 f32 -> lives in 192MB L2)
  for (int idx = lane; idx < ME*OD; idx += 32) {
    int r = idx / OD, d = idx - r*OD;
    atomicAdd(&nsum[s_src[r]*HD + d], s_out[r*HD + d]);
  }
}

// segment mean + equivariant layer norm + residual; also re-zeros sums
template<int L>
__global__ void node_kernel(float* __restrict__ nsum, const float* __restrict__ cnt,
                            const float* __restrict__ gamma, const float* __restrict__ beta,
                            float* __restrict__ h) {
  int n = blockIdx.x * blockDim.x + threadIdx.x;
  if (n >= NN) return;
  constexpr int OD = Cfg<L>::OD;
  float c = cnt[n]; c = c < 1.f ? 1.f : c;
  float inv = 1.f / c;
  float u[HD];
  for (int d = 0; d < OD; ++d) { u[d] = nsum[n*HD + d] * inv; nsum[n*HD + d] = 0.f; }
  int off = 0, ch = 0;
#pragma unroll
  for (int ir = 0; ir < Cfg<L>::NIR; ++ir) {
    const int m  = Cfg<L>::IRR[ir][0];
    const int l  = Cfg<L>::IRR[ir][1];
    const int p  = Cfg<L>::IRR[ir][2];
    const int dd = 2*l + 1;
    if (l == 0 && p == 1) {
      float mu = 0.f;
      for (int i = 0; i < m; ++i) mu += u[off + i];
      mu /= m;
      float var = 0.f;
      for (int i = 0; i < m; ++i) { float t = u[off + i] - mu; var += t*t; }
      var /= m;
      float rs = rsqrtf(var + EPSV);
      for (int i = 0; i < m; ++i)
        u[off + i] = (u[off + i] - mu) * rs * gamma[ch + i] + beta[ch + i];
    } else {
      float n2 = 0.f;
      for (int i = 0; i < m; ++i)
        for (int k = 0; k < dd; ++k) { float t = u[off + i*dd + k]; n2 += t*t; }
      n2 /= m;
      float rs = rsqrtf(n2 + EPSV);
      for (int i = 0; i < m; ++i)
        for (int k = 0; k < dd; ++k) u[off + i*dd + k] *= rs * gamma[ch + i];
    }
    off += m*dd; ch += m;
  }
  for (int d = 0; d < OD; ++d) h[n*HD + d] += u[d];
}

__global__ void ffn_kernel(const float* __restrict__ h,
                           const float* __restrict__ w1, const float* __restrict__ b1,
                           const float* __restrict__ w2, const float* __restrict__ b2,
                           float* __restrict__ out) {
  int n = blockIdx.x * blockDim.x + threadIdx.x;
  if (n >= NN) return;
  float e[32], t[32];
#pragma unroll
  for (int i = 0; i < 16; ++i) { e[i] = h[n*HD + i]; e[16 + i] = h[n*HD + 64 + i]; }
  for (int o = 0; o < 32; ++o) {
    float a = b1[o];
#pragma unroll
    for (int j = 0; j < 32; ++j) a += e[j] * w1[j*32 + o];
    t[o] = a > 0.f ? a : 0.f;
  }
  for (int o = 0; o < 16; ++o) {
    float a = b2[o];
#pragma unroll
    for (int j = 0; j < 32; ++j) a += t[j] * w2[j*16 + o];
    out[n*16 + o] = a;
  }
}

// ---------------------------------------------------------------------------
extern "C" void kernel_launch(void* const* d_in, const int* in_sizes, int n_in,
                              void* d_out, int out_size, void* d_ws, size_t ws_size,
                              hipStream_t stream) {
  (void)in_sizes; (void)n_in; (void)out_size; (void)ws_size;
  // inputs in setup_inputs() insertion order; params dict flattened in order
  const float* node_attr = (const float*)d_in[0];
  const float* coords    = (const float*)d_in[1];
  const int*   edges     = (const int*)  d_in[3];   // [2] batch_id unused
  const float* eattr     = (const float*)d_in[4];
  const float* emb_w1 = (const float*)d_in[5], *emb_b1 = (const float*)d_in[6];
  const float* emb_w2 = (const float*)d_in[7], *emb_b2 = (const float*)d_in[8];
  const float *fw1[4], *fb1[4], *fw2[4], *fb2[4], *gam[4], *bet[4];
  for (int l = 0; l < 4; ++l) {
    int b = 9 + 6*l;
    fw1[l] = (const float*)d_in[b+0]; fb1[l] = (const float*)d_in[b+1];
    fw2[l] = (const float*)d_in[b+2]; fb2[l] = (const float*)d_in[b+3];
    gam[l] = (const float*)d_in[b+4]; bet[l] = (const float*)d_in[b+5];
  }
  const float* ffn_w1 = (const float*)d_in[33], *ffn_b1 = (const float*)d_in[34];
  const float* ffn_w2 = (const float*)d_in[35], *ffn_b2 = (const float*)d_in[36];

  // workspace carve (all offsets 32B aligned)
  char* p = (char*)d_ws;
  float* shG = (float*)p; p += (size_t)NEG*9*4;
  float* emG = (float*)p; p += (size_t)NEG*32*4;
  float* hG  = (float*)p; p += (size_t)NN*HD*4;
  float* smG = (float*)p; p += (size_t)NN*HD*4;
  float* ctG = (float*)p; p += (size_t)NN*4;
  static const int NT2s[4] = {24, 44, 64, 86};
  static const int WNs[4]  = {384, 704, 1024, 1368};
  unsigned short *w1pG[4], *w2pG[4];
  for (int l = 0; l < 4; ++l) {
    w1pG[l] = (unsigned short*)p; p += (size_t)3*6*512*2;
    w2pG[l] = (unsigned short*)p; p += (size_t)3*NT2s[l]*512*2;
  }

  init_kernel<<<(NN*HD + 255)/256, 256, 0, stream>>>(node_attr, hG, smG, ctG);
  geom_kernel<<<(NEG + 255)/256, 256, 0, stream>>>(coords, edges, emb_w1, emb_b1,
                                                   emb_w2, emb_b2, shG, emG, ctG);
  for (int l = 0; l < 4; ++l) {
    int t1 = 3*6*512;
    pack_kernel<<<(t1 + 255)/256, 256, 0, stream>>>(fw1[l], w1pG[l], 96, 96, 6);
    int t2 = 3*NT2s[l]*512;
    pack_kernel<<<(t2 + 255)/256, 256, 0, stream>>>(fw2[l], w2pG[l], 96, WNs[l], NT2s[l]);
  }

  constexpr int EB = NEG / 32;   // 2500 single-wave blocks x 32 edges
  constexpr int NB = (NN + 255)/256;

  edge_kernel<0><<<EB, 32, 0, stream>>>(edges, emG, eattr, shG, hG, w1pG[0], w2pG[0],
                                        fb1[0], fb2[0], smG);
  node_kernel<0><<<NB, 256, 0, stream>>>(smG, ctG, gam[0], bet[0], hG);
  edge_kernel<1><<<EB, 32, 0, stream>>>(edges, emG, eattr, shG, hG, w1pG[1], w2pG[1],
                                        fb1[1], fb2[1], smG);
  node_kernel<1><<<NB, 256, 0, stream>>>(smG, ctG, gam[1], bet[1], hG);
  edge_kernel<2><<<EB, 32, 0, stream>>>(edges, emG, eattr, shG, hG, w1pG[2], w2pG[2],
                                        fb1[2], fb2[2], smG);
  node_kernel<2><<<NB, 256, 0, stream>>>(smG, ctG, gam[2], bet[2], hG);
  edge_kernel<3><<<EB, 32, 0, stream>>>(edges, emG, eattr, shG, hG, w1pG[3], w2pG[3],
                                        fb1[3], fb2[3], smG);
  node_kernel<3><<<NB, 256, 0, stream>>>(smG, ctG, gam[3], bet[3], hG);

  ffn_kernel<<<NB, 256, 0, stream>>>(hG, ffn_w1, ffn_b1, ffn_w2, ffn_b2, (float*)d_out);
}
